// ScaledDotProductAttention_20933670600908
// MI455X (gfx1250) — matrix operand order repaired and checked by reading, not compiled, not run on hardware
//
#include <hip/hip_runtime.h>

// Problem constants (from reference): B=4, H=16, S=2048, D=128
#define BSZ 4
#define HN 16
#define SEQ 2048
#define DIM 128
#define TK 64            // keys per KV block
#define SCALE 0.08838834764831843f   // 128^-0.5

#define KH_ELEMS ((size_t)BSZ * HN * SEQ * DIM)   // 16,777,216 f16 elems per tensor
#define KH_BYTES (KH_ELEMS * 2)                   // 33,554,432 bytes

typedef _Float16 v16h __attribute__((ext_vector_type(16)));
typedef _Float16 v8h  __attribute__((ext_vector_type(8)));
typedef _Float16 v4h  __attribute__((ext_vector_type(4)));
typedef float    v8f  __attribute__((ext_vector_type(8)));
typedef unsigned int v4u __attribute__((ext_vector_type(4)));
typedef int      v8i  __attribute__((ext_vector_type(8)));
typedef int      v4i  __attribute__((ext_vector_type(4)));

#if __has_builtin(__builtin_amdgcn_tensor_load_to_lds) && \
    __has_builtin(__builtin_amdgcn_s_wait_tensorcnt)
#define USE_TDM 1
#else
#define USE_TDM 0
#endif

// Build a 16-element f16 WMMA fragment from two contiguous 16-byte LDS chunks
// (lo -> VGPRs 0..3, hi -> VGPRs 4..7 per CDNA5 A/B-matrix layouts).
static __device__ __forceinline__ v16h ldfrag(const _Float16* lo, const _Float16* hi) {
    v8h a = *(const v8h*)lo;
    v8h b = *(const v8h*)hi;
    v16h r;
#pragma unroll
    for (int i = 0; i < 8; ++i) { r[i] = a[i]; r[i + 8] = b[i]; }
    return r;
}

#if USE_TDM
// Issue one TDM 2D tile load: rows x cols f16 elements, global row stride = cols
// of the source tensor (tensor_dim0_stride), into LDS at lds_dst.
static __device__ __forceinline__ void tdm_load_f16(const _Float16* gsrc, void* lds_dst,
                                                    int tile_d0, int tile_d1, int stride0) {
    unsigned long long ga = (unsigned long long)(uintptr_t)gsrc;
    v4u g0;
    g0[0] = 1u;                                   // count=1 (valid user D#)
    g0[1] = (unsigned int)(uintptr_t)lds_dst;     // LDS byte address
    g0[2] = (unsigned int)ga;                     // global_addr[31:0]
    g0[3] = ((unsigned int)(ga >> 32) & 0x01FFFFFFu) | 0x80000000u; // addr[56:32], type=2
    v8i g1;
    unsigned td0 = 1u << 20, td1 = 1u << 20;      // huge tensor dims: no OOB clipping
    g1[0] = 0x10000;                              // data_size=1 (2 bytes)
    g1[1] = (int)((td0 & 0xFFFFu) << 16);         // tensor_dim0 lo16
    g1[2] = (int)((td0 >> 16) | ((td1 & 0xFFFFu) << 16));   // dim0 hi | dim1 lo
    g1[3] = (int)((td1 >> 16) | ((unsigned)tile_d0 << 16)); // dim1 hi | tile_dim0
    g1[4] = tile_d1;                              // tile_dim1 (0 = 1D)
    g1[5] = stride0;                              // tensor_dim0_stride (elements)
    g1[6] = 0;
    g1[7] = 0;
    v4i zz4 = { 0, 0, 0, 0 };
    v8i zz8 = { 0, 0, 0, 0, 0, 0, 0, 0 };
    __builtin_amdgcn_tensor_load_to_lds(g0, g1, zz4, zz4, zz8, 0);
}
#endif

// ---- Pre-pass: K -> f16 (row-major), V -> f16 transposed per 64-key block ----
// Vth layout: [bh][blk][d][key%64] so a KV block is one contiguous 16 KB chunk.
__global__ __launch_bounds__(256)
void cvt_kv_kernel(const float* __restrict__ K, const float* __restrict__ V,
                   _Float16* __restrict__ Kh, _Float16* __restrict__ Vth) {
    size_t qi = (size_t)blockIdx.x * 256 + threadIdx.x;   // quad index
    size_t e  = qi * 4;                                   // element index
    size_t bh  = e / ((size_t)SEQ * DIM);
    size_t rem = e % ((size_t)SEQ * DIM);
    int s = (int)(rem / DIM);
    int d = (int)(rem % DIM);

    float4 kf = *(const float4*)(K + e);
    v4h kh = { (_Float16)kf.x, (_Float16)kf.y, (_Float16)kf.z, (_Float16)kf.w };
    *(v4h*)(Kh + e) = kh;

    float4 vf = *(const float4*)(V + e);
    _Float16* vb = Vth + bh * ((size_t)SEQ * DIM) + (size_t)(s >> 6) * (DIM * TK)
                       + (size_t)d * TK + (s & 63);
    vb[0 * TK] = (_Float16)vf.x;
    vb[1 * TK] = (_Float16)vf.y;
    vb[2 * TK] = (_Float16)vf.z;
    vb[3 * TK] = (_Float16)vf.w;
}

template <bool WS>
__global__ __launch_bounds__(128)
void fa_fwd_kernel(const float* __restrict__ Q, const float* __restrict__ K,
                   const float* __restrict__ V, const int* __restrict__ mask,
                   float* __restrict__ out,
                   const _Float16* __restrict__ Kh, const _Float16* __restrict__ Vth) {
    // LDS: K tile (f16 row-major [key][d]), V tile transposed (f16 [d][key]),
    // per-wave P scratch (f16 [q][key]).  40 KB total.
    __shared__ _Float16 Ksh[TK * DIM];        // 16 KB
    __shared__ _Float16 Vt[DIM * TK];         // 16 KB
    __shared__ _Float16 Psh[4][16 * TK];      // 8 KB

    const int tid  = threadIdx.x;
    const int wave = tid >> 5;
    const int lane = tid & 31;
    const int hi16 = (lane >> 4) & 1;   // lane half (0: lanes 0-15, 1: lanes 16-31)
    const int ln   = lane & 15;

    const int bh = blockIdx.y;          // b*H + h
    const int b  = bh >> 4;             // H == 16
    const int qw = blockIdx.x * 64 + wave * 16;  // this wave's first query row

    const float* Qg = Q + (size_t)bh * SEQ * DIM;
    const float* Kg = K + (size_t)bh * SEQ * DIM;
    const float* Vg = V + (size_t)bh * SEQ * DIM;
    const int*   Mg = mask + (size_t)b * SEQ * SEQ;
    float*       Og = out + (size_t)bh * SEQ * DIM;

    // ---- Q B-fragments (k = d, n = query), resident for the whole kernel ----
    v16h qf[4];
    {
        const float* qrow = Qg + (size_t)(qw + ln) * DIM;
#pragma unroll
        for (int ch = 0; ch < 4; ++ch) {
            const float4* p = (const float4*)(qrow + ch * 32 + hi16 * 16);
#pragma unroll
            for (int j = 0; j < 4; ++j) {
                float4 f = p[j];
                qf[ch][j * 4 + 0] = (_Float16)f.x;
                qf[ch][j * 4 + 1] = (_Float16)f.y;
                qf[ch][j * 4 + 2] = (_Float16)f.z;
                qf[ch][j * 4 + 3] = (_Float16)f.w;
            }
        }
    }

    v8f o[8] = {};                 // O accumulators: 8 d-tiles of 16x16
    float m_run = -1e30f;          // running row max
    float l_run = 0.0f;            // running row sum

    for (int blk = 0; blk < SEQ / TK; ++blk) {
        const int k0 = blk * TK;
        __syncthreads();

        if constexpr (WS) {
#if USE_TDM
            // ---- both KV tiles via Tensor Data Mover (pre-converted f16) ----
            if (wave == 0) {
                // K tile: 2D, 128 f16 per row x 64 key rows, row stride 128
                tdm_load_f16(Kh + (size_t)bh * SEQ * DIM + (size_t)k0 * DIM,
                             &Ksh[0], DIM, TK, DIM);
                // V^T tile: pre-transposed, one contiguous 8192-element chunk
                tdm_load_f16(Vth + (size_t)bh * SEQ * DIM + (size_t)k0 * DIM,
                             &Vt[0], DIM * TK, 0, DIM * TK);
                __builtin_amdgcn_s_wait_tensorcnt(0);
            }
#else
            // Workspace present but no TDM builtins: plain f16 block copies.
#pragma unroll
            for (int i = 0; i < 8; ++i) {
                int c8 = tid + i * 128;   // 1024 x 16B chunks per 16 KB tile
                *(v8h*)(&Ksh[c8 * 8]) =
                    *(const v8h*)(Kh + (size_t)bh * SEQ * DIM + (size_t)k0 * DIM + c8 * 8);
                *(v8h*)(&Vt[c8 * 8]) =
                    *(const v8h*)(Vth + (size_t)bh * SEQ * DIM + (size_t)k0 * DIM + c8 * 8);
            }
#endif
        } else {
            // ---- no workspace: stage f32 -> f16 manually (K row-major, V^T) ----
#pragma unroll
            for (int i = 0; i < 16; ++i) {
                int idx = tid + i * 128;        // 0..2047 quads
                int row = idx >> 5;             // key row 0..63
                int dc  = (idx & 31) << 2;      // d 0..124 step 4
                float4 kf = *(const float4*)(Kg + (size_t)(k0 + row) * DIM + dc);
                v4h kh = { (_Float16)kf.x, (_Float16)kf.y, (_Float16)kf.z, (_Float16)kf.w };
                *(v4h*)(&Ksh[row * DIM + dc]) = kh;
                float4 vf = *(const float4*)(Vg + (size_t)(k0 + row) * DIM + dc);
                Vt[(dc + 0) * TK + row] = (_Float16)vf.x;
                Vt[(dc + 1) * TK + row] = (_Float16)vf.y;
                Vt[(dc + 2) * TK + row] = (_Float16)vf.z;
                Vt[(dc + 3) * TK + row] = (_Float16)vf.w;
            }
        }
        __syncthreads();

        // ---- S^T = K_tile (64x128) x Q^T (128x16): 4 M-tiles x 4 k-chunks ----
        v8f c[4] = {};
#pragma unroll
        for (int t = 0; t < 4; ++t) {
#pragma unroll
            for (int ch = 0; ch < 4; ++ch) {
                // A layout: m = ln (key), k = ch*32 + hi16*16 + {0..15}
                const _Float16* ab = &Ksh[(t * 16 + ln) * DIM + ch * 32 + hi16 * 16];
                v16h a = ldfrag(ab, ab + 8);
                c[t] = __builtin_amdgcn_wmma_f32_16x16x32_f16(
                           false, a, false, qf[ch], (short)0, c[t], false, false);
            }
        }

        // ---- scale, clip, mask (lane: query q = qw+ln, keys t*16+hi16*8+r) ----
#pragma unroll
        for (int t = 0; t < 4; ++t) {
            const int* mrow = Mg + (size_t)(qw + ln) * SEQ + k0 + t * 16 + hi16 * 8;
            int4 m0 = *(const int4*)(mrow);
            int4 m1 = *(const int4*)(mrow + 4);
            int mv[8] = { m0.x, m0.y, m0.z, m0.w, m1.x, m1.y, m1.z, m1.w };
#pragma unroll
            for (int r = 0; r < 8; ++r) {
                float lg = c[t][r] * SCALE;
                lg = fminf(fmaxf(lg, -10.0f), 10.0f);
                c[t][r] = (mv[r] == 0) ? -1e9f : lg;
            }
        }

        // ---- online softmax (stats per query live in lanes ln and ln+16) ----
        float bmax = -3e38f;
#pragma unroll
        for (int t = 0; t < 4; ++t)
#pragma unroll
            for (int r = 0; r < 8; ++r) bmax = fmaxf(bmax, c[t][r]);
        bmax = fmaxf(bmax, __shfl_xor(bmax, 16));
        float m_new  = fmaxf(m_run, bmax);
        float sc_old = __expf(m_run - m_new);
        float bsum = 0.0f;
#pragma unroll
        for (int t = 0; t < 4; ++t) {
            v8h ph;
#pragma unroll
            for (int r = 0; r < 8; ++r) {
                float e = __expf(c[t][r] - m_new);
                bsum += e;
                ph[r] = (_Float16)e;
            }
            // P[q][key] f16, wave-private scratch (LDS in-order within a wave)
            *(v8h*)(&Psh[wave][ln * TK + t * 16 + hi16 * 8]) = ph;
        }
        bsum += __shfl_xor(bsum, 16);
        l_run = l_run * sc_old + bsum;
        m_run = m_new;

        // ---- rescale O accumulators (O layout: q = r + hi16*8 per VGPR r) ----
#pragma unroll
        for (int r = 0; r < 8; ++r) {
            float s = __shfl(sc_old, r + hi16 * 8);
#pragma unroll
            for (int dt = 0; dt < 8; ++dt) o[dt][r] *= s;
        }

        // ---- O += P (16x64) x V (64x128): P A-fragments + Vt B-fragments ----
        v16h pf[2];
#pragma unroll
        for (int ch = 0; ch < 2; ++ch) {
            const _Float16* pb = &Psh[wave][ln * TK + ch * 32 + hi16 * 8];
            pf[ch] = ldfrag(pb, pb + 16);
        }
#pragma unroll
        for (int dt = 0; dt < 8; ++dt) {
#pragma unroll
            for (int ch = 0; ch < 2; ++ch) {
                // B layout: n = d = dt*16 + ln, k = ch*32 + hi16*8 + {0..7, 16..23} keys
                const _Float16* vb = &Vt[(dt * 16 + ln) * TK + ch * 32 + hi16 * 8];
                v16h bf = ldfrag(vb, vb + 16);
                o[dt] = __builtin_amdgcn_wmma_f32_16x16x32_f16(
                            false, pf[ch], false, bf, (short)0, o[dt], false, false);
            }
        }
    }

    // ---- finalize: O / l, write out (C layout: q = r + hi16*8, d = dt*16 + ln) ----
#pragma unroll
    for (int r = 0; r < 8; ++r) {
        float li  = __shfl(l_run, r + hi16 * 8);
        float inv = 1.0f / li;
        float* orow = Og + (size_t)(qw + r + hi16 * 8) * DIM + ln;
#pragma unroll
        for (int dt = 0; dt < 8; ++dt) {
            orow[dt * 16] = o[dt][r] * inv;
        }
    }
}

extern "C" void kernel_launch(void* const* d_in, const int* in_sizes, int n_in,
                              void* d_out, int out_size, void* d_ws, size_t ws_size,
                              hipStream_t stream) {
    const float* Q    = (const float*)d_in[0];
    const float* K    = (const float*)d_in[1];
    const float* V    = (const float*)d_in[2];
    const int*   mask = (const int*)d_in[3];
    float*       out  = (float*)d_out;

    dim3 grid(SEQ / 64, BSZ * HN);   // 32 x 64 workgroups
    dim3 block(128);                 // 4 waves; each wave owns 16 query rows

    if (d_ws != nullptr && ws_size >= 2 * KH_BYTES) {
        _Float16* Kh  = (_Float16*)d_ws;
        _Float16* Vth = (_Float16*)((char*)d_ws + KH_BYTES);
        // Pre-pass: convert K/V to f16 (V pre-transposed per 64-key block).
        cvt_kv_kernel<<<dim3((unsigned)(KH_ELEMS / 4 / 256)), dim3(256), 0, stream>>>(
            K, V, Kh, Vth);
        fa_fwd_kernel<true><<<grid, block, 0, stream>>>(Q, K, V, mask, out, Kh, Vth);
    } else {
        fa_fwd_kernel<false><<<grid, block, 0, stream>>>(Q, K, V, mask, out, nullptr, nullptr);
    }
}
